// SemanticCritic_30691836297666
// MI455X (gfx1250) — matrix operand order repaired and checked by reading, not compiled, not run on hardware
//
#include <hip/hip_runtime.h>

typedef __attribute__((ext_vector_type(16))) _Float16 v16h;
typedef __attribute__((ext_vector_type(8)))  _Float16 v8h;
typedef __attribute__((ext_vector_type(8)))  float    v8f;
typedef __attribute__((ext_vector_type(4)))  int      v4i;

// explicit address-space types so weight/bias traffic lowers to global_load_*
typedef __attribute__((address_space(1))) float    g_float;
typedef __attribute__((address_space(1))) v16h     g_v16h;
typedef __attribute__((address_space(1))) v8h      g_v8h;
typedef __attribute__((address_space(1))) v4i      g_v4i;
typedef __attribute__((address_space(3))) v4i      l_v4i;

#define B_TOTAL 65536

// workspace layout, in halves (f16)
#define OFF_W1T 0
#define OFF_W3T 16384
#define OFF_W2T 32768
#define OFF_W4T (32768 + 65536)
#define OFF_W5T (32768 + 2*65536)
#define OFF_W7T (32768 + 3*65536)
#define WS_HALVES (32768 + 4*65536)   // 294912 halves = 576 KB

// padded LDS stride (halves) to spread fragment loads across all 64 banks
#define STR_H  264    // 16x256 H tiles and 256x256 weight tiles

// ---------------------------------------------------------------------------
// f32 -> f16 transposed (+zero-padded) weight images, N-major: Wt[n*K + k]
// so a WMMA B-fragment is one contiguous 32-byte run per lane.
// ---------------------------------------------------------------------------
__global__ __launch_bounds__(256)
void prep_weights(const float* __restrict__ w1, const float* __restrict__ w3,
                  const float* __restrict__ w2, const float* __restrict__ w4,
                  const float* __restrict__ w5, const float* __restrict__ w7,
                  _Float16* __restrict__ ws)
{
    int idx = blockIdx.x * 256 + threadIdx.x;
    if (idx >= WS_HALVES) return;
    float v;
    if (idx < 32768) {                       // w1t / w3t : 256 x 64 (K padded 54->64)
        const float* src = (idx < 16384) ? w1 : w3;
        int r = idx & 16383;
        int n = r >> 6, k = r & 63;
        v = (k < 54) ? *(const g_float*)(src + k * 256 + n) : 0.0f;
    } else {                                 // w2t/w4t/w5t/w7t : 256 x 256
        int j = idx - 32768;
        int which = j >> 16;
        int r = j & 65535;
        int n = r >> 8, k = r & 255;
        const float* src = (which == 0) ? w2 : (which == 1) ? w4 : (which == 2) ? w5 : w7;
        v = *(const g_float*)(src + k * 256 + n);
    }
    ws[idx] = (_Float16)v;
}

// ---------------------------------------------------------------------------
// Fragment loaders: contiguous k order, element s of half hi is k = k0 + 16*hi + s (A and B match).
// ---------------------------------------------------------------------------
__device__ __forceinline__ v16h load_a_lds(const _Float16* base, int stride,
                                           int l16, int hi, int k0)
{
    const v8h lo = *(const v8h*)(base + l16 * stride + k0 + hi * 16);
    const v8h hh = *(const v8h*)(base + l16 * stride + k0 + hi * 16 + 8);
    return __builtin_shufflevector(lo, hh, 0,1,2,3,4,5,6,7,8,9,10,11,12,13,14,15);
}

__device__ __forceinline__ v16h load_b_global(const _Float16* wt, int K,
                                              int col, int k0, int hi)
{
    return *(const g_v16h*)(wt + col * K + k0 + hi * 16);
}

__device__ __forceinline__ v16h load_b_lds(const _Float16* w, int col, int k0, int hi)
{
    const _Float16* p = w + col * STR_H + k0 + hi * 16;
    const v8h lo = *(const v8h*)(p);
    const v8h hh = *(const v8h*)(p + 8);
    return __builtin_shufflevector(lo, hh, 0,1,2,3,4,5,6,7,8,9,10,11,12,13,14,15);
}

__device__ __forceinline__ float ldg(const float* p) { return *(const g_float*)p; }

__device__ __forceinline__ void wait_async0()
{
#if __has_builtin(__builtin_amdgcn_s_wait_asynccnt)
    __builtin_amdgcn_s_wait_asynccnt(0);
#else
    asm volatile("s_wait_asynccnt 0x0" ::: "memory");
#endif
}

// Stage a 256x256 f16 weight image (K-major rows) into LDS with row stride
// STR_H, via the gfx1250 async global->LDS path.
__device__ __forceinline__ void stage_weight_256(const _Float16* __restrict__ src,
                                                 _Float16* dstLds, int tid)
{
#if __has_builtin(__builtin_amdgcn_global_load_async_to_lds_b128)
    for (int c = tid; c < 8192; c += 256) {          // 8192 chunks of 8 halves
        int n = c >> 5;
        int w = (c & 31) * 8;
        __builtin_amdgcn_global_load_async_to_lds_b128(
            (g_v4i*)(src + n * 256 + w),
            (l_v4i*)(dstLds + n * STR_H + w),
            0, 0);
    }
    wait_async0();
#else
    for (int c = tid; c < 8192; c += 256) {
        int n = c >> 5;
        int w = (c & 31) * 8;
        *(v8h*)(dstLds + n * STR_H + w) = *(const g_v8h*)(src + n * 256 + w);
    }
#endif
}

// gather one element of the pair-input row (combinations_trick layout)
__device__ __forceinline__ float in_val(int c, int p, int bit2, int oi, int oj,
                                        const float* obsR, const float* agR,
                                        const float* gR,   const float* actR)
{
    float val;
    if      (c == 0) val = ldg(agR + p);
    else if (c == 1) val = ldg(agR + bit2);
    else if (c == 2) val = ldg(gR + p);
    else if (c == 3) val = ldg(gR + bit2);
    else if (c < 14) val = ldg(obsR + (c - 4));
    else if (c < 32) { int t = c - 14;
        val = (t < 3) ? ((t == oi) ? 1.0f : 0.0f)
                      : ldg(obsR + 10 + 15 * oi + (t - 3)); }
    else if (c < 50) { int t = c - 32;
        val = (t < 3) ? ((t == oj) ? 1.0f : 0.0f)
                      : ldg(obsR + 10 + 15 * oj + (t - 3)); }
    else if (c < 54) val = ldg(actR + (c - 50));
    else             val = 0.0f;
    return val;
}

// ---------------------------------------------------------------------------
// Fused critic. 256 threads = 8 waves, 16 rows/wave, 128 rows/block.
// Per MLP: stage W2 in LDS (shared, async) -> 3x (build layer-1 A-fragments
// in registers, layer1 WMMA from global W1t, layer2 WMMA from LDS,
// relu-accumulate S in regs) -> stage W5 -> rho.
// ---------------------------------------------------------------------------
__global__ __launch_bounds__(256)
void critic_fused(const float* __restrict__ obs, const float* __restrict__ ag,
                  const float* __restrict__ g,   const float* __restrict__ anchor_g,
                  const float* __restrict__ act,
                  const float* __restrict__ pb1, const float* __restrict__ pb2,
                  const float* __restrict__ pb3, const float* __restrict__ pb4,
                  const float* __restrict__ pb5, const float* __restrict__ pb7,
                  const float* __restrict__ w6,  const float* __restrict__ pb6,
                  const float* __restrict__ w8,  const float* __restrict__ pb8,
                  const _Float16* __restrict__ wsh,
                  float* __restrict__ out)
{
    __shared__ __align__(16) _Float16 lds_h[8 * 16 * STR_H];   // 66 KB
    __shared__ __align__(16) _Float16 lds_w[256 * STR_H];      // 132 KB

    const int tid  = threadIdx.x;
    const int wave = tid >> 5;
    const int lane = tid & 31;
    const int l16  = lane & 15;
    const int hi   = lane >> 4;
    const int rowBase = blockIdx.x * 128 + wave * 16;

    _Float16* myH = lds_h + wave * (16 * STR_H);

    __builtin_prefetch(wsh + OFF_W1T, 0, 3);
    __builtin_prefetch(wsh + OFF_W3T, 0, 3);

    const int P_o1[3] = {0, 0, 1};
    const int P_o2[3] = {1, 2, 2};
    const int P_j[3]  = {3, 4, 6};
    const int P_k[3]  = {5, 7, 8};

    const int row = rowBase + l16;
    const float* obsR = obs + row * 55;
    const float* agR  = ag  + row * 9;
    const float* gR   = g   + row * 9;
    const float* actR = act + row * 4;

    const _Float16* WA[2]   = { wsh + OFF_W1T, wsh + OFF_W3T };
    const _Float16* WBm[2]  = { wsh + OFF_W2T, wsh + OFF_W4T };
    const _Float16* WC[2]   = { wsh + OFF_W5T, wsh + OFF_W7T };
    const float*    BL1[2]  = { pb1, pb3 };
    const float*    BL2[2]  = { pb2, pb4 };
    const float*    BL3[2]  = { pb5, pb7 };
    const float*    WOUT[2] = { w6,  w8  };
    const float*    BOUT[2] = { pb6, pb8 };

    #pragma unroll 1
    for (int m = 0; m < 2; ++m) {
        // S accumulator lives in registers: 16 N-tiles x v8f
        v8f sacc[16];
        #pragma unroll
        for (int nt = 0; nt < 16; ++nt) { v8f z = {}; sacc[nt] = z; }

        __syncthreads();                        // previous lds_w consumers done
        stage_weight_256(WBm[m], lds_w, tid);   // W2t -> LDS (async)
        __syncthreads();

        #pragma unroll 1
        for (int p = 0; p < 3; ++p) {
            // ---- build this pair's two layer-1 A-fragments in registers ----
            v16h a0, a1;
            {
                const int o1 = P_o1[p], o2 = P_o2[p], j = P_j[p], k = P_k[p];
                const bool sel = (ldg(anchor_g + row * 9 + j) -
                                  ldg(anchor_g + row * 9 + k)) >= 0.0f;
                const int bit2 = sel ? j  : k;
                const int oi   = sel ? o1 : o2;
                const int oj   = sel ? o2 : o1;
                #pragma unroll 1
                for (int jj = 0; jj < 16; ++jj) {
                    const int cb = hi * 16 + jj;
                    a0[jj] = (_Float16)in_val(cb,      p, bit2, oi, oj,
                                              obsR, agR, gR, actR);
                    a1[jj] = (_Float16)in_val(32 + cb, p, bit2, oi, oj,
                                              obsR, agR, gR, actR);
                }
            }

            // ---- layer 1: H = relu(inp @ W1 + b1), K=64, B from global ----
            #pragma unroll 1
            for (int nt = 0; nt < 16; ++nt) {
                v8f c = {};
                v16h b0 = load_b_global(WA[m], 64, nt * 16 + l16, 0, hi);
                c = __builtin_amdgcn_wmma_f32_16x16x32_f16(false, a0, false, b0,
                                                           (short)0, c, false, false);
                v16h b1 = load_b_global(WA[m], 64, nt * 16 + l16, 32, hi);
                c = __builtin_amdgcn_wmma_f32_16x16x32_f16(false, a1, false, b1,
                                                           (short)0, c, false, false);
                const float bias = ldg(BL1[m] + nt * 16 + l16);
                #pragma unroll
                for (int r = 0; r < 8; ++r) {
                    float h = c[r] + bias;
                    h = h > 0.f ? h : 0.f;
                    myH[(r + 8 * hi) * STR_H + nt * 16 + l16] = (_Float16)h;
                }
            }

            // ---- layer 2: S += relu(H @ W2 + b2), K=256, B from LDS -------
            #pragma unroll
            for (int nt = 0; nt < 16; ++nt) {
                v8f c = {};
                #pragma unroll 2
                for (int kt = 0; kt < 8; ++kt) {
                    v16h a  = load_a_lds(myH, STR_H, l16, hi, kt * 32);
                    v16h bb = load_b_lds(lds_w, nt * 16 + l16, kt * 32, hi);
                    c = __builtin_amdgcn_wmma_f32_16x16x32_f16(false, a, false, bb,
                                                               (short)0, c, false, false);
                }
                const float bias = ldg(BL2[m] + nt * 16 + l16);
                #pragma unroll
                for (int r = 0; r < 8; ++r) {
                    float x = c[r] + bias;
                    x = x > 0.f ? x : 0.f;
                    sacc[nt][r] += x;
                }
            }
        }

        __syncthreads();                        // everyone done reading W2t
        stage_weight_256(WC[m], lds_w, tid);    // W5t -> LDS (async)

        // while staging completes, spill S (f32 regs) -> f16 A-tile in myH
        #pragma unroll
        for (int nt = 0; nt < 16; ++nt) {
            #pragma unroll
            for (int r = 0; r < 8; ++r)
                myH[(r + 8 * hi) * STR_H + nt * 16 + l16] = (_Float16)sacc[nt][r];
        }
        __syncthreads();

        // ---- rho: q = relu(S @ W5 + b5) @ w6 + b6, fused -------------------
        float qp[8];
        #pragma unroll
        for (int r = 0; r < 8; ++r) qp[r] = 0.f;

        #pragma unroll 1
        for (int nt = 0; nt < 16; ++nt) {
            v8f c = {};
            #pragma unroll 2
            for (int kt = 0; kt < 8; ++kt) {
                v16h a  = load_a_lds(myH, STR_H, l16, hi, kt * 32);
                v16h bb = load_b_lds(lds_w, nt * 16 + l16, kt * 32, hi);
                c = __builtin_amdgcn_wmma_f32_16x16x32_f16(false, a, false, bb,
                                                           (short)0, c, false, false);
            }
            const float bias = ldg(BL3[m] + nt * 16 + l16);
            const float wv   = ldg(WOUT[m] + nt * 16 + l16);
            #pragma unroll
            for (int r = 0; r < 8; ++r) {
                float u = c[r] + bias;
                u = u > 0.f ? u : 0.f;
                qp[r] += u * wv;
            }
        }

        // reduce 16-lane partial dots; lane l16==0 of each half holds rows
        // r (hi=0) and r+8 (hi=1)
        const float bout = ldg(BOUT[m]);
        #pragma unroll
        for (int r = 0; r < 8; ++r) {
            float v = qp[r];
            v += __shfl_xor(v, 1, 16);
            v += __shfl_xor(v, 2, 16);
            v += __shfl_xor(v, 4, 16);
            v += __shfl_xor(v, 8, 16);
            if (l16 == 0)
                *(g_float*)(out + m * B_TOTAL + rowBase + r + 8 * hi) = v + bout;
        }
    }
}

extern "C" void kernel_launch(void* const* d_in, const int* in_sizes, int n_in,
                              void* d_out, int out_size, void* d_ws, size_t ws_size,
                              hipStream_t stream)
{
    const float* obs      = (const float*)d_in[0];
    const float* ag       = (const float*)d_in[1];
    const float* g        = (const float*)d_in[2];
    const float* anchor_g = (const float*)d_in[3];
    const float* act      = (const float*)d_in[4];
    const float* w1 = (const float*)d_in[5];  const float* b1 = (const float*)d_in[6];
    const float* w2 = (const float*)d_in[7];  const float* b2 = (const float*)d_in[8];
    const float* w3 = (const float*)d_in[9];  const float* b3 = (const float*)d_in[10];
    const float* w4 = (const float*)d_in[11]; const float* b4 = (const float*)d_in[12];
    const float* w5 = (const float*)d_in[13]; const float* b5 = (const float*)d_in[14];
    const float* w6 = (const float*)d_in[15]; const float* b6 = (const float*)d_in[16];
    const float* w7 = (const float*)d_in[17]; const float* b7 = (const float*)d_in[18];
    const float* w8 = (const float*)d_in[19]; const float* b8 = (const float*)d_in[20];

    _Float16* wsh = (_Float16*)d_ws;
    float*    out = (float*)d_out;

    prep_weights<<<dim3((WS_HALVES + 255) / 256), dim3(256), 0, stream>>>(
        w1, w3, w2, w4, w5, w7, wsh);

    critic_fused<<<dim3(B_TOTAL / 128), dim3(256), 0, stream>>>(
        obs, ag, g, anchor_g, act,
        b1, b2, b3, b4, b5, b7,
        w6, b6, w8, b8,
        wsh, out);
}
